// EPDOModel_73813307949509
// MI455X (gfx1250) — matrix operand order repaired and checked by reading, not compiled, not run on hardware
//
#include <hip/hip_runtime.h>
#include <hip/hip_bf16.h>
#include <math.h>

// ---------------------------------------------------------------------------
// Problem constants (match reference)
// ---------------------------------------------------------------------------
#define NN 16384
#define NE 262144
#define HID 128
#define NBLK 15
#define EPSV 1e-5f

typedef __bf16 bf16;
typedef bf16  v16bf __attribute__((ext_vector_type(16)));
typedef float v8f   __attribute__((ext_vector_type(8)));
typedef unsigned int u32;

// ---------------------------------------------------------------------------
// bf16 helpers (round-to-nearest-even, bit level)
// ---------------------------------------------------------------------------
__device__ __forceinline__ u32 f2bf_u(float f) {
    u32 u = __float_as_uint(f);
    return (u + 0x7FFFu + ((u >> 16) & 1u)) >> 16;
}
__device__ __forceinline__ bf16 f2bf(float f) {
    unsigned short s = (unsigned short)f2bf_u(f);
    bf16 b; __builtin_memcpy(&b, &s, 2); return b;
}
__device__ __forceinline__ float bf2f(bf16 b) {
    unsigned short s; __builtin_memcpy(&s, &b, 2);
    return __uint_as_float(((u32)s) << 16);
}
__device__ __forceinline__ u32 pack2bf(float a, float b) {
    return f2bf_u(a) | (f2bf_u(b) << 16);
}

// ---------------------------------------------------------------------------
// Generic bf16 WMMA GEMM:  Y[M,N] = A[M,Kpad] @ Wpacked[Kpad,N] + bias
//   AMODE 0: A is a dense bf16 matrix with leading dim lda (padding pre-zeroed)
//   AMODE 1: A row e = [edgeE[e] | nodeA[dst[e]] | nodeA[src[e]]]  (K = 384)
//   AMODE 2: A row r = [nodeA[r] | bf16(aggF[r])]                  (K = 256)
//   EPI 0: write bf16 Y, accumulate per-column BN stats (sum, sumsq) -> stats
//   EPI 1: write bf16 Y only
//   EPI 2: write f32 Y only
// NACC = accumulators per wave (compile-time): N = NACC * 32.
// Block = 256 threads = 8 wave32; tile 64(M) x N; K stepped by 32 via LDS.
// All B fragments for a K-step are preloaded into registers (one clause of
// global_load_b128), then NACC independent v_wmma issue back-to-back.
// ---------------------------------------------------------------------------
template <int AMODE, int NACC>
__global__ __launch_bounds__(256) void gemm_bf16_wmma(
    const bf16* __restrict__ A, int lda,
    const bf16* __restrict__ edgeE, const bf16* __restrict__ nodeA,
    const float* __restrict__ aggF,
    const int* __restrict__ dstIdx, const int* __restrict__ srcIdx,
    const bf16* __restrict__ Wp, const float* __restrict__ bias, int Nbias,
    int M, int Kpad, int EPI,
    bf16* __restrict__ Ybf, float* __restrict__ Yf, float* __restrict__ stats)
{
    constexpr int N     = NACC * 32;
    constexpr int numNT = N >> 4;      // 16-col tiles across N

    __shared__ bf16  As[64][48];       // 32 K-cols + pad (16B-aligned rows)
    __shared__ float sred[2 * N];      // per-column BN partials (sum | sumsq)

    const int tid   = threadIdx.x;
    const int mBase = blockIdx.x * 64;
    const int KT    = Kpad >> 5;

    if (tid < 2 * N) sred[tid] = 0.f;

    const int wave   = tid >> 5;
    const int lane   = tid & 31;
    const int mw     = (wave & 3) * 16;        // wave's M offset in tile
    const int ntBase = (wave >> 2) * NACC;     // wave's first N-tile

    v8f acc[NACC];
#pragma unroll
    for (int t = 0; t < NACC; ++t) acc[t] = (v8f){0.f,0.f,0.f,0.f,0.f,0.f,0.f,0.f};

    // cooperative A-tile fill assignment: 64 rows x 4 chunks of 8 bf16
    const int row = tid >> 2;
    const int kc  = (tid & 3) * 8;
    const int gm  = mBase + row;

    // Hoist loop-invariant gather rows for the edge-concat mode
    const bf16* rowEdge = nullptr;
    const bf16* rowDst  = nullptr;
    const bf16* rowSrc  = nullptr;
    if constexpr (AMODE == 1) {
        if (gm < M) {
            rowEdge = edgeE + (size_t)gm * HID;
            rowDst  = nodeA + (size_t)dstIdx[gm] * HID;
            rowSrc  = nodeA + (size_t)srcIdx[gm] * HID;
        }
    }

    const v16bf* wv = (const v16bf*)Wp;

    for (int kt = 0; kt < KT; ++kt) {
        __syncthreads();
        const int gk = kt * 32 + kc;
        uint4 t4 = make_uint4(0u, 0u, 0u, 0u);
        if (gm < M) {
            if constexpr (AMODE == 0) {
                t4 = *(const uint4*)(A + (size_t)gm * lda + gk);
            } else if constexpr (AMODE == 1) {
                const int seg = gk >> 7, off = gk & 127;
                const bf16* base = (seg == 0) ? rowEdge : (seg == 1) ? rowDst : rowSrc;
                t4 = *(const uint4*)(base + off);
            } else {
                if (gk < 128) {
                    t4 = *(const uint4*)(nodeA + (size_t)gm * HID + gk);
                } else {
                    const float* fp = aggF + (size_t)gm * HID + (gk - 128);
                    float4 f0 = *(const float4*)(fp);
                    float4 f1 = *(const float4*)(fp + 4);
                    t4.x = pack2bf(f0.x, f0.y); t4.y = pack2bf(f0.z, f0.w);
                    t4.z = pack2bf(f1.x, f1.y); t4.w = pack2bf(f1.z, f1.w);
                }
            }
        }
        *(uint4*)(&As[row][kc]) = t4;

        // Preload ALL B fragments for this K-step (independent of LDS barrier)
        union { v16bf v; uint4 q[2]; } bf[NACC];
#pragma unroll
        for (int t = 0; t < NACC; ++t) {
            const uint4* bp =
                (const uint4*)(wv + ((size_t)(kt * numNT + ntBase + t) * 32 + lane));
            bf[t].q[0] = bp[0];
            bf[t].q[1] = bp[1];
        }
        if (kt + 1 < KT)  // prefetch next K-step's weight fragments
            __builtin_prefetch(wv + ((size_t)((kt + 1) * numNT + ntBase) * 32 + lane), 0, 0);

        __syncthreads();

        // A fragment: contiguous K order (matches packed B)
        //   lanes 0-15 : row m, K = 0..7 (q0) and 8..15 (q1)
        //   lanes 16-31: row m, K = 16..23 (q0) and 24..31 (q1)
        const int am = lane & 15, ah = lane >> 4;
        const bf16* arow = &As[mw + am][0];
        union { v16bf v; uint4 q[2]; } af;
        af.q[0] = *(const uint4*)(arow + ah * 16);
        af.q[1] = *(const uint4*)(arow + ah * 16 + 8);

#pragma unroll
        for (int t = 0; t < NACC; ++t)
            acc[t] = __builtin_amdgcn_wmma_f32_16x16x32_bf16(
                false, af.v, false, bf[t].v, (short)0, acc[t], false, false);
    }

    __syncthreads();

    // Epilogue. C layout: lane&15 = column-in-tile, (lane>>4)*8 + r = row.
#pragma unroll
    for (int t = 0; t < NACC; ++t) {
        const int nt  = ntBase + t;
        const int col = nt * 16 + (lane & 15);
        const float bv = (bias && col < Nbias) ? bias[col] : 0.f;
        float s = 0.f, s2 = 0.f;
#pragma unroll
        for (int r = 0; r < 8; ++r) {
            const int grow = mBase + mw + ((lane >> 4) * 8) + r;
            if (grow < M) {
                const float y = acc[t][r] + bv;
                if (EPI == 2) Yf[(size_t)grow * N + col] = y;
                else          Ybf[(size_t)grow * N + col] = f2bf(y);
                if (EPI == 0) { s += y; s2 += y * y; }
            }
        }
        if (EPI == 0) {
            s  += __shfl_xor(s,  16, 32);   // lanes l and l+16 share a column
            s2 += __shfl_xor(s2, 16, 32);
            if (lane < 16) {
                atomicAdd(&sred[col],     s);
                atomicAdd(&sred[N + col], s2);
            }
        }
    }
    if (EPI == 0) {
        __syncthreads();
        if (tid < 2 * N) atomicAdd(&stats[tid], sred[tid]);
    }
}

// ---------------------------------------------------------------------------
// BatchNorm (column stats over M rows) + exact GELU, bf16 in/out
// ---------------------------------------------------------------------------
__global__ void bn_gelu_kernel(const bf16* __restrict__ Y, bf16* __restrict__ Out,
                               const float* __restrict__ stats,
                               const float* __restrict__ g, const float* __restrict__ b,
                               int M, int N)
{
    size_t i = (size_t)blockIdx.x * blockDim.x + threadIdx.x;
    if (i >= (size_t)M * N) return;
    const int col = (int)(i % N);
    const float invM = 1.f / (float)M;
    const float mean = stats[col] * invM;
    const float var  = stats[N + col] * invM - mean * mean;
    const float x = (bf2f(Y[i]) - mean) * rsqrtf(var + EPSV) * g[col] + b[col];
    Out[i] = f2bf(0.5f * x * (1.f + erff(x * 0.70710678118654752f)));
}

// ---------------------------------------------------------------------------
// Row LayerNorm over 128 columns, one wave32 per row (4 cols/lane)
// ---------------------------------------------------------------------------
__device__ __forceinline__ void ln_row(const bf16* __restrict__ yrow,
                                       const float* __restrict__ g,
                                       const float* __restrict__ b,
                                       float out[4], int lane)
{
    float x[4];
    float s = 0.f;
#pragma unroll
    for (int j = 0; j < 4; ++j) { x[j] = bf2f(yrow[lane + 32 * j]); s += x[j]; }
#pragma unroll
    for (int o = 16; o >= 1; o >>= 1) s += __shfl_xor(s, o, 32);
    const float m = s * (1.f / 128.f);
    float ss = 0.f;
#pragma unroll
    for (int j = 0; j < 4; ++j) { float d = x[j] - m; ss += d * d; }
#pragma unroll
    for (int o = 16; o >= 1; o >>= 1) ss += __shfl_xor(ss, o, 32);
    const float inv = rsqrtf(ss * (1.f / 128.f) + EPSV);
#pragma unroll
    for (int j = 0; j < 4; ++j) {
        const int c = lane + 32 * j;
        out[j] = (x[j] - m) * inv * g[c] + b[c];
    }
}

// edge-block LN: write bf16 edge state + fused segment_sum into aggF (f32)
__global__ void ln_edge_agg_kernel(const bf16* __restrict__ Y,
                                   const float* __restrict__ g, const float* __restrict__ b,
                                   bf16* __restrict__ edgeE, float* __restrict__ aggF,
                                   const int* __restrict__ dstIdx, int M)
{
    const int lane = threadIdx.x & 31;
    const int row  = blockIdx.x * 8 + (threadIdx.x >> 5);
    if (row >= M) return;
    float o[4];
    ln_row(Y + (size_t)row * HID, g, b, o, lane);
    const int d = dstIdx[row];
#pragma unroll
    for (int j = 0; j < 4; ++j) {
        const int c = lane + 32 * j;
        edgeE[(size_t)row * HID + c] = f2bf(o[j]);
        atomicAdd(&aggF[(size_t)d * HID + c], o[j]);
    }
}

// plain LN -> bf16 (edge encoder)
__global__ void ln_plain_kernel(const bf16* __restrict__ Y,
                                const float* __restrict__ g, const float* __restrict__ b,
                                bf16* __restrict__ out, int M)
{
    const int lane = threadIdx.x & 31;
    const int row  = blockIdx.x * 8 + (threadIdx.x >> 5);
    if (row >= M) return;
    float o[4];
    ln_row(Y + (size_t)row * HID, g, b, o, lane);
#pragma unroll
    for (int j = 0; j < 4; ++j)
        out[(size_t)row * HID + lane + 32 * j] = f2bf(o[j]);
}

// node encoder LN -> f32 master + bf16 copy
__global__ void ln_node_init_kernel(const bf16* __restrict__ Y,
                                    const float* __restrict__ g, const float* __restrict__ b,
                                    float* __restrict__ nodeF, bf16* __restrict__ nodeA, int M)
{
    const int lane = threadIdx.x & 31;
    const int row  = blockIdx.x * 8 + (threadIdx.x >> 5);
    if (row >= M) return;
    float o[4];
    ln_row(Y + (size_t)row * HID, g, b, o, lane);
#pragma unroll
    for (int j = 0; j < 4; ++j) {
        const size_t idx = (size_t)row * HID + lane + 32 * j;
        nodeF[idx] = o[j];
        nodeA[idx] = f2bf(o[j]);
    }
}

// node block tail: LN + residual + per-graph branch scaling, f32 master + bf16
__global__ void ln_resid_scale_kernel(const bf16* __restrict__ Y,
                                      const float* __restrict__ g, const float* __restrict__ b,
                                      const float* __restrict__ bout,   // [2 x 128]
                                      float* __restrict__ nodeF, bf16* __restrict__ nodeA, int M)
{
    const int lane = threadIdx.x & 31;
    const int row  = blockIdx.x * 8 + (threadIdx.x >> 5);
    if (row >= M) return;
    float o[4];
    ln_row(Y + (size_t)row * HID, g, b, o, lane);
    const int graph = row >> 13;   // NN / N_GRAPHS = 8192
#pragma unroll
    for (int j = 0; j < 4; ++j) {
        const int c = lane + 32 * j;
        const size_t idx = (size_t)row * HID + c;
        const float nf = (nodeF[idx] + o[j]) * bout[graph * HID + c];
        nodeF[idx] = nf;
        nodeA[idx] = f2bf(nf);
    }
}

// ---------------------------------------------------------------------------
// Weight packing: f32 [Kreal x Nreal] -> bf16 fragment layout for WMMA B
// tile (kt,nt) of 32x16; within tile lane-major, lane's 16 values are its
// K-run (kbase = (lane>>4)*16) for column n = lane&15.
// ---------------------------------------------------------------------------
__global__ void pack_w_kernel(const float* __restrict__ src, bf16* __restrict__ dst,
                              int Kreal, int Kpad, int Nreal, int Npad)
{
    const int total = Kpad * Npad;
    const int i = blockIdx.x * 256 + threadIdx.x;
    if (i >= total) return;
    const int numNT = Npad >> 4;
    const int tile = i >> 9;           // 512 elems per tile
    const int rem  = i & 511;
    const int lane = rem >> 4, j = rem & 15;
    const int kt = tile / numNT, nt = tile % numNT;
    const int n  = lane & 15, kb = (lane >> 4) * 16;
    const int k   = kt * 32 + kb + j;
    const int col = nt * 16 + n;
    const float v = (k < Kreal && col < Nreal) ? src[(size_t)k * Nreal + col] : 0.f;
    dst[i] = f2bf(v);
}

// ---------------------------------------------------------------------------
// Input staging: build zero-padded bf16 A matrices (K padded to 32)
// ---------------------------------------------------------------------------
__global__ void build_node_in_kernel(const float* __restrict__ fd,
                                     const float* __restrict__ oh, bf16* __restrict__ out)
{
    const int i = blockIdx.x * 256 + threadIdx.x;
    if (i >= NN * 32) return;
    const int n = i >> 5, c = i & 31;
    float v = 0.f;
    if (c < 3)       v = fd[n * 6 + c * 2 + 1];      // field_data[:, c, 1]
    else if (c < 12) v = oh[n * 9 + (c - 3)];
    out[i] = f2bf(v);
}

__global__ void build_edge_in_kernel(const float* __restrict__ ef, bf16* __restrict__ out)
{
    const int i = blockIdx.x * 256 + threadIdx.x;
    if (i >= NE * 32) return;
    const int e = i >> 5, c = i & 31;
    out[i] = f2bf(c < 3 ? ef[e * 3 + c] : 0.f);
}

__global__ void zero_f32_kernel(float* __restrict__ p, long long n)
{
    const long long i = (long long)blockIdx.x * 256 + threadIdx.x;
    if (i < n) p[i] = 0.f;
}

// ---------------------------------------------------------------------------
// Tiny branch MLP (2 graphs): 2->40->128->128 silu, ->128 linear
// ---------------------------------------------------------------------------
__device__ __forceinline__ float silu(float x) { return x / (1.f + __expf(-x)); }

__global__ void branch_kernel(const float* __restrict__ bin,
                              const float* __restrict__ W1, const float* __restrict__ b1,
                              const float* __restrict__ W2, const float* __restrict__ b2,
                              const float* __restrict__ W3, const float* __restrict__ b3,
                              const float* __restrict__ W4, const float* __restrict__ b4,
                              float* __restrict__ bout)
{
    __shared__ float h1[2][40];
    __shared__ float h2[2][128];
    __shared__ float h3[2][128];
    const int t = threadIdx.x;   // 128 threads
    for (int gph = 0; gph < 2; ++gph)
        if (t < 40) {
            float a = b1[t] + bin[gph * 2 + 0] * W1[0 * 40 + t]
                            + bin[gph * 2 + 1] * W1[1 * 40 + t];
            h1[gph][t] = silu(a);
        }
    __syncthreads();
    for (int gph = 0; gph < 2; ++gph) {
        float a = b2[t];
        for (int k = 0; k < 40; ++k) a += h1[gph][k] * W2[k * 128 + t];
        h2[gph][t] = silu(a);
    }
    __syncthreads();
    for (int gph = 0; gph < 2; ++gph) {
        float a = b3[t];
        for (int k = 0; k < 128; ++k) a += h2[gph][k] * W3[k * 128 + t];
        h3[gph][t] = silu(a);
    }
    __syncthreads();
    for (int gph = 0; gph < 2; ++gph) {
        float a = b4[t];
        for (int k = 0; k < 128; ++k) a += h3[gph][k] * W4[k * 128 + t];
        bout[gph * 128 + t] = a;
    }
}

// ---------------------------------------------------------------------------
// Output: residual + bundle multiplier; out shape (NN, 3, 10) f32
// ---------------------------------------------------------------------------
__global__ void final_kernel(const float* __restrict__ fd,
                             const float* __restrict__ dec, float* __restrict__ out)
{
    const int i = blockIdx.x * 256 + threadIdx.x;
    if (i >= NN * 30) return;
    const int n = i / 30, q = i % 30, f = q / 10, j = q % 10;
    out[i] = fd[n * 6 + f * 2 + 1] + dec[n * 32 + f * 10 + j] * (float)(j + 1);
}

// ---------------------------------------------------------------------------
// Host orchestration
// ---------------------------------------------------------------------------
#define MB(x) ((size_t)(x) * 1024ull * 1024ull)

extern "C" void kernel_launch(void* const* d_in, const int* in_sizes, int n_in,
                              void* d_out, int out_size, void* d_ws, size_t ws_size,
                              hipStream_t stream)
{
    (void)in_sizes; (void)n_in; (void)out_size; (void)ws_size;
    auto F = [&](int i) { return (const float*)d_in[i]; };

    const float* fd   = F(0);
    const float* oh   = F(1);
    const int*   eidx = (const int*)d_in[2];
    const int*   srcIdx = eidx;        // edge_index[0]
    const int*   dstIdx = eidx + NE;   // edge_index[1]
    const float* ef   = F(3);
    const float* bin  = F(4);

    char* ws = (char*)d_ws;
    bf16*  wsPack  = (bf16*)(ws + 0);          // packed weights (~4.6 MB)
    float* nodeF   = (float*)(ws + MB(16));    // f32 node master   8 MB
    bf16*  nodeA   = (bf16*)(ws + MB(24));     // bf16 node state   4 MB
    bf16*  node_in = (bf16*)(ws + MB(28));     // NN x 32           1 MB
    float* aggF    = (float*)(ws + MB(29));    // f32 aggregation   8 MB
    bf16*  yN      = (bf16*)(ws + MB(37));     // node pre-act      4 MB
    bf16*  aN      = (bf16*)(ws + MB(41));     // node act          4 MB
    float* decbuf  = (float*)(ws + MB(45));    // NN x 32 f32       2 MB
    float* bout    = (float*)(ws + MB(47));    // 2 x 128
    float* stats   = (float*)(ws + MB(48));    // 256 f32
    bf16*  edge_in = (bf16*)(ws + MB(49));     // NE x 32          16 MB
    bf16*  edgeE   = (bf16*)(ws + MB(65));     // NE x 128         64 MB
    bf16*  yE      = (bf16*)(ws + MB(129));    // NE x 128         64 MB
    bf16*  aE      = (bf16*)(ws + MB(193));    // NE x 128         64 MB

    // ---- packed weight offsets ----
    size_t po = 0;
    auto palloc = [&](size_t e) { size_t r = po; po += e; return r; };
    const size_t o_neW1 = palloc(32 * 128),  o_neW2 = palloc(16384), o_neW3 = palloc(16384);
    const size_t o_eeW1 = palloc(32 * 128),  o_eeW2 = palloc(16384), o_eeW3 = palloc(16384);
    const size_t o_dW1  = palloc(16384),     o_dW2  = palloc(16384), o_dW3  = palloc(128 * 32);
    const size_t o_beW1 = palloc((size_t)NBLK * 49152);
    const size_t o_beW2 = palloc((size_t)NBLK * 16384);
    const size_t o_beW3 = palloc((size_t)NBLK * 16384);
    const size_t o_bnW1 = palloc((size_t)NBLK * 32768);
    const size_t o_bnW2 = palloc((size_t)NBLK * 16384);
    const size_t o_bnW3 = palloc((size_t)NBLK * 16384);

    auto pack = [&](const float* src, size_t off, int Kr, int Kp, int Nr, int Np) {
        const int tot = Kp * Np;
        pack_w_kernel<<<(tot + 255) / 256, 256, 0, stream>>>(src, wsPack + off, Kr, Kp, Nr, Np);
    };
    auto zerof = [&](float* p, long long n) {
        zero_f32_kernel<<<(unsigned)((n + 255) / 256), 256, 0, stream>>>(p, n);
    };

    // N = 128 dense GEMM
    auto gemm0 = [&](const bf16* A, int lda, size_t wOff, const float* bias,
                     int M, int Kp, int EPI, bf16* Yb, float* st) {
        gemm_bf16_wmma<0, 4><<<(M + 63) / 64, 256, 0, stream>>>(
            A, lda, nullptr, nullptr, nullptr, nullptr, nullptr,
            wsPack + wOff, bias, HID, M, Kp, EPI, Yb, nullptr, st);
    };
    // N = 32 dense GEMM (decoder final, f32 out)
    auto gemm0n32 = [&](const bf16* A, size_t wOff, const float* bias, int Nbias,
                        float* Yfp) {
        gemm_bf16_wmma<0, 1><<<(NN + 63) / 64, 256, 0, stream>>>(
            A, 128, nullptr, nullptr, nullptr, nullptr, nullptr,
            wsPack + wOff, bias, Nbias, NN, 128, 2, nullptr, Yfp, nullptr);
    };
    auto gemm1 = [&](size_t wOff, const float* bias, bf16* Yb, float* st) {
        gemm_bf16_wmma<1, 4><<<(NE + 63) / 64, 256, 0, stream>>>(
            nullptr, 0, edgeE, nodeA, nullptr, dstIdx, srcIdx,
            wsPack + wOff, bias, HID, NE, 384, 0, Yb, nullptr, st);
    };
    auto gemm2 = [&](size_t wOff, const float* bias, bf16* Yb, float* st) {
        gemm_bf16_wmma<2, 4><<<(NN + 63) / 64, 256, 0, stream>>>(
            nullptr, 0, nullptr, nodeA, aggF, nullptr, nullptr,
            wsPack + wOff, bias, HID, NN, 256, 0, Yb, nullptr, st);
    };
    auto bngelu = [&](const bf16* Y, bf16* O, const float* g, const float* b, int M) {
        const long long tot = (long long)M * HID;
        bn_gelu_kernel<<<(unsigned)((tot + 255) / 256), 256, 0, stream>>>(Y, O, stats, g, b, M, HID);
    };

    // ---- 0) pack all weights ----
    pack(F(5),  o_neW1, 12, 32, 128, 128);
    pack(F(9),  o_neW2, 128, 128, 128, 128);
    pack(F(13), o_neW3, 128, 128, 128, 128);
    pack(F(17), o_eeW1, 3, 32, 128, 128);
    pack(F(21), o_eeW2, 128, 128, 128, 128);
    pack(F(25), o_eeW3, 128, 128, 128, 128);
    pack(F(29), o_dW1, 128, 128, 128, 128);
    pack(F(33), o_dW2, 128, 128, 128, 128);
    pack(F(37), o_dW3, 128, 128, 30, 32);
    for (int i = 0; i < NBLK; ++i) {
        pack(F(39) + (size_t)i * 384 * 128, o_beW1 + (size_t)i * 49152, 384, 384, 128, 128);
        pack(F(43) + (size_t)i * 128 * 128, o_beW2 + (size_t)i * 16384, 128, 128, 128, 128);
        pack(F(47) + (size_t)i * 128 * 128, o_beW3 + (size_t)i * 16384, 128, 128, 128, 128);
        pack(F(51) + (size_t)i * 256 * 128, o_bnW1 + (size_t)i * 32768, 256, 256, 128, 128);
        pack(F(55) + (size_t)i * 128 * 128, o_bnW2 + (size_t)i * 16384, 128, 128, 128, 128);
        pack(F(59) + (size_t)i * 128 * 128, o_bnW3 + (size_t)i * 16384, 128, 128, 128, 128);
    }

    // ---- 1) staging + branch net ----
    build_node_in_kernel<<<(NN * 32 + 255) / 256, 256, 0, stream>>>(fd, oh, node_in);
    build_edge_in_kernel<<<(NE * 32 + 255) / 256, 256, 0, stream>>>(ef, edge_in);
    branch_kernel<<<1, 128, 0, stream>>>(bin, F(63), F(64), F(65), F(66),
                                         F(67), F(68), F(69), F(70), bout);

    // ---- 2) node encoder ----
    zerof(stats, 256);
    gemm0(node_in, 32, o_neW1, F(6), NN, 32, 0, yN, stats);
    bngelu(yN, aN, F(7), F(8), NN);
    zerof(stats, 256);
    gemm0(aN, 128, o_neW2, F(10), NN, 128, 0, yN, stats);
    bngelu(yN, aN, F(11), F(12), NN);
    gemm0(aN, 128, o_neW3, F(14), NN, 128, 1, yN, nullptr);
    ln_node_init_kernel<<<NN / 8, 256, 0, stream>>>(yN, F(15), F(16), nodeF, nodeA, NN);

    // ---- 3) edge encoder ----
    zerof(stats, 256);
    gemm0(edge_in, 32, o_eeW1, F(18), NE, 32, 0, yE, stats);
    bngelu(yE, aE, F(19), F(20), NE);
    zerof(stats, 256);
    gemm0(aE, 128, o_eeW2, F(22), NE, 128, 0, yE, stats);
    bngelu(yE, aE, F(23), F(24), NE);
    gemm0(aE, 128, o_eeW3, F(26), NE, 128, 1, yE, nullptr);
    ln_plain_kernel<<<NE / 8, 256, 0, stream>>>(yE, F(27), F(28), edgeE, NE);

    // ---- 4) message-passing blocks ----
    for (int i = 0; i < NBLK; ++i) {
        const size_t i128 = (size_t)i * 128;
        // edge MLP (gathered concat input)
        zerof(aggF, (long long)NN * HID);
        zerof(stats, 256);
        gemm1(o_beW1 + (size_t)i * 49152, F(40) + i128, yE, stats);
        bngelu(yE, aE, F(41) + i128, F(42) + i128, NE);
        zerof(stats, 256);
        gemm0(aE, 128, o_beW2 + (size_t)i * 16384, F(44) + i128, NE, 128, 0, yE, stats);
        bngelu(yE, aE, F(45) + i128, F(46) + i128, NE);
        gemm0(aE, 128, o_beW3 + (size_t)i * 16384, F(48) + i128, NE, 128, 1, yE, nullptr);
        // LN + fused segment_sum into aggF
        ln_edge_agg_kernel<<<NE / 8, 256, 0, stream>>>(yE, F(49) + i128, F(50) + i128,
                                                       edgeE, aggF, dstIdx, NE);
        // node MLP (gathered [node | agg] input)
        zerof(stats, 256);
        gemm2(o_bnW1 + (size_t)i * 32768, F(52) + i128, yN, stats);
        bngelu(yN, aN, F(53) + i128, F(54) + i128, NN);
        zerof(stats, 256);
        gemm0(aN, 128, o_bnW2 + (size_t)i * 16384, F(56) + i128, NN, 128, 0, yN, stats);
        bngelu(yN, aN, F(57) + i128, F(58) + i128, NN);
        gemm0(aN, 128, o_bnW3 + (size_t)i * 16384, F(60) + i128, NN, 128, 1, yN, nullptr);
        ln_resid_scale_kernel<<<NN / 8, 256, 0, stream>>>(yN, F(61) + i128, F(62) + i128,
                                                          bout, nodeF, nodeA, NN);
    }

    // ---- 5) decoder (no LN) ----
    zerof(stats, 256);
    gemm0(nodeA, 128, o_dW1, F(30), NN, 128, 0, yN, stats);
    bngelu(yN, aN, F(31), F(32), NN);
    zerof(stats, 256);
    gemm0(aN, 128, o_dW2, F(34), NN, 128, 0, yN, stats);
    bngelu(yN, aN, F(35), F(36), NN);
    gemm0n32(aN, o_dW3, F(38), 30, decbuf);

    // ---- 6) output ----
    final_kernel<<<(NN * 30 + 255) / 256, 256, 0, stream>>>(fd, decbuf, (float*)d_out);
}
